// SwinTransformerBlock_66889820668240
// MI455X (gfx1250) — matrix operand order repaired and checked by reading, not compiled, not run on hardware
//
#include <hip/hip_runtime.h>
#include <cstddef>
#include <cstdint>

typedef __attribute__((ext_vector_type(16))) _Float16 v16h;
typedef __attribute__((ext_vector_type(8)))  _Float16 v8h;
typedef __attribute__((ext_vector_type(8)))  float    v8f;

#define DIM   384
#define NHEAD 12
#define HD    32
#define WS    7
#define SS_   3
#define NTOK  49
#define NPAD  64
#define H_    28
#define W_    28
#define HID   1536
#define SCALE 0.17677669529663687f   /* 32^-0.5 */

union ABu { v16h v; v8h h8[2]; };

__device__ __forceinline__ v8f vzero8() {
  v8f z;
#pragma unroll
  for (int i = 0; i < 8; ++i) z[i] = 0.f;
  return z;
}

__device__ __forceinline__ v8f wmma16(v16h a, v16h b, v8f c) {
  // D = A(16x32 f16) * B(32x16 f16) + C(16x16 f32)
  return __builtin_amdgcn_wmma_f32_16x16x32_f16(false, a, false, b, (short)0, c, false, false);
}

// Async global->LDS copy (CDNA5, tracked by ASYNCcnt). vdst = wave-relative LDS
// byte address; generic pointers to __shared__ have the LDS offset in the low
// 32 bits (flat LDS aperture truncation).
__device__ __forceinline__ void async_g2l_b128(void* lds_dst, const void* gsrc) {
  unsigned l = (unsigned)(uintptr_t)lds_dst;
  asm volatile("global_load_async_to_lds_b128 %0, %1, off"
               :: "v"(l), "v"(gsrc) : "memory");
}
__device__ __forceinline__ void wait_async0() {
  asm volatile("s_wait_asynccnt 0x0" ::: "memory");
}

// A fragment: lane = M (lane&15); lanes 0-15 hold K {0..7,16..23}, lanes 16-31 hold K {8..15,24..31}
__device__ __forceinline__ v16h ldA(const _Float16* base, int ld) {
  const unsigned lane = threadIdx.x & 31u;
  const _Float16* r = base + (size_t)(lane & 15u) * ld + ((lane & 16u) ? 8 : 0);
  ABu u;
  u.h8[0] = *(const v8h*)(r);
  u.h8[1] = *(const v8h*)(r + 16);
  return u.v;
}

// B fragment from K-transposed storage (row n holds all K contiguously):
// lane = N (lane&15); lanes 0-15 hold K {0..7,16..23}, lanes 16-31 hold K {8..15,24..31}
__device__ __forceinline__ v16h ldB(const _Float16* base, int ld) {
  const unsigned lane = threadIdx.x & 31u;
  const _Float16* r = base + (size_t)(lane & 15u) * ld + ((lane & 16u) ? 8 : 0);
  ABu u;
  u.h8[0] = *(const v8h*)(r);
  u.h8[1] = *(const v8h*)(r + 16);
  return u.v;
}

// D store to fp16 tile (row-major): lane = N, VGPR v = row (mb+v)
__device__ __forceinline__ void stTile(_Float16* base, int ld, v8f c, float bias, float scale) {
  const unsigned lane = threadIdx.x & 31u;
  const int n = lane & 15;
  const int mb = (lane & 16u) ? 8 : 0;
#pragma unroll
  for (int v = 0; v < 8; ++v)
    base[(size_t)(mb + v) * ld + n] = (_Float16)((c[v] + bias) * scale);
}

__device__ __forceinline__ int regid(int u) { return u < (H_ - WS) ? 0 : (u < (H_ - SS_) ? 1 : 2); }

// ---------------------------------------------------------------------------
// Kernel 0: weight conversion (fp32 -> fp16, transposed) + rel-bias table
// ---------------------------------------------------------------------------
__global__ void swin_prep_kernel(const float* __restrict__ qkv_w, const float* __restrict__ proj_w,
                                 const float* __restrict__ w1,    const float* __restrict__ w2,
                                 const float* __restrict__ rel_bias,
                                 _Float16* __restrict__ qkvT, _Float16* __restrict__ projT,
                                 _Float16* __restrict__ w1T,  _Float16* __restrict__ w2T,
                                 float* __restrict__ biasTab) {
  const int S0 = 1152 * 384;
  const int S1 = S0 + 384 * 384;
  const int S2 = S1 + 1536 * 384;
  const int S3 = S2 + 384 * 1536;
  const int S4 = S3 + NHEAD * NPAD * NPAD;
  int idx = blockIdx.x * 256 + threadIdx.x;
  if (idx < S0) {
    int n = idx / 384, k = idx % 384;
    qkvT[idx] = (_Float16)qkv_w[(size_t)k * 1152 + n];
  } else if (idx < S1) {
    int j = idx - S0; int n = j / 384, k = j % 384;
    projT[j] = (_Float16)proj_w[(size_t)k * 384 + n];
  } else if (idx < S2) {
    int j = idx - S1; int n = j / 384, k = j % 384;
    w1T[j] = (_Float16)w1[(size_t)k * 1536 + n];
  } else if (idx < S3) {
    int j = idx - S2; int n = j / 1536, k = j % 1536;
    w2T[j] = (_Float16)w2[(size_t)k * 384 + n];
  } else if (idx < S4) {
    int j = idx - S3;
    int h = j >> 12; int r = (j >> 6) & 63; int c = j & 63;
    float v = 0.f;
    if (r < NTOK && c < NTOK) {
      int di = r / WS - c / WS + (WS - 1);
      int dj = r % WS - c % WS + (WS - 1);
      v = rel_bias[(size_t)(di * (2 * WS - 1) + dj) * NHEAD + h];
    }
    biasTab[j] = v;
  }
}

// ---------------------------------------------------------------------------
// Kernel 1: fused LN1 + shifted-window gather + QKV + attention + proj + res
// One workgroup (8 wave32) per window. Writes x1 = x + o into `out`.
// ---------------------------------------------------------------------------
__global__ __launch_bounds__(256)
void swin_attn_kernel(const float* __restrict__ x,
                      const float* __restrict__ ln1_g, const float* __restrict__ ln1_b,
                      const float* __restrict__ qkv_b, const float* __restrict__ proj_b,
                      const _Float16* __restrict__ qkvT, const _Float16* __restrict__ projT,
                      const float* __restrict__ biasTab,
                      float* __restrict__ out) {
  __shared__ __align__(16) _Float16 sYO[NPAD * DIM];        // Y (LN'd input), later O
  __shared__ __align__(16) _Float16 sQ[NPAD * DIM];
  __shared__ __align__(16) _Float16 sK[NPAD * DIM];
  __shared__ __align__(16) _Float16 sVt[DIM * NPAD];        // V transposed: [dim][token]
  __shared__ __align__(16) _Float16 sPW[8 * 16 * NPAD];     // per-wave softmax staging

  const int b  = blockIdx.x >> 4;
  const int wi = blockIdx.x & 15;
  const int wy = wi >> 2, wx = wi & 3;
  const int wv = threadIdx.x >> 5;
  const unsigned lane = threadIdx.x & 31u;
  const int n15 = lane & 15;
  const int mb  = (lane & 16u) ? 8 : 0;

  // ---- Phase 0a: async-DMA the shifted-window gather of x into LDS staging
  //      (sQ/sK region is dead until phase 1) ----
  float* sX = (float*)sQ;                      // 49*384 fp32 = 75 KB staging
  for (int i = threadIdx.x; i < NTOK * (DIM / 4); i += 256) {
    const int t = i / (DIM / 4);
    const int q = i - t * (DIM / 4);
    int ti = t / WS, tj = t - ti * WS;
    int hh = wy * WS + ti + SS_; if (hh >= H_) hh -= H_;
    int ww = wx * WS + tj + SS_; if (ww >= W_) ww -= W_;
    const float* g = x + ((size_t)b * (H_ * W_) + hh * W_ + ww) * DIM + q * 4;
    async_g2l_b128(sX + (size_t)t * DIM + q * 4, g);
  }
  for (int p = threadIdx.x; p < (NPAD - NTOK) * DIM; p += 256)
    sYO[NTOK * DIM + p] = (_Float16)0.f;
  wait_async0();
  __syncthreads();

  // ---- Phase 0b: LN1 from LDS staging into fp16 Y ----
  {
    const int c0 = lane * 12;
    float g[12], be[12];
#pragma unroll
    for (int k = 0; k < 12; ++k) { g[k] = ln1_g[c0 + k]; be[k] = ln1_b[c0 + k]; }
    for (int t = wv; t < NTOK; t += 8) {
      const float* xr = sX + (size_t)t * DIM + c0;
      float vb[12];
#pragma unroll
      for (int q = 0; q < 3; ++q) {
        float4 f = *(const float4*)(xr + q * 4);
        vb[q * 4 + 0] = f.x; vb[q * 4 + 1] = f.y; vb[q * 4 + 2] = f.z; vb[q * 4 + 3] = f.w;
      }
      float s = 0.f, ss = 0.f;
#pragma unroll
      for (int k = 0; k < 12; ++k) { s += vb[k]; ss += vb[k] * vb[k]; }
#pragma unroll
      for (int off = 16; off; off >>= 1) {
        s  += __shfl_xor(s,  off, 32);
        ss += __shfl_xor(ss, off, 32);
      }
      float mean = s * (1.f / DIM);
      float rstd = rsqrtf(ss * (1.f / DIM) - mean * mean + 1e-5f);
#pragma unroll
      for (int k = 0; k < 12; ++k)
        sYO[(size_t)t * DIM + c0 + k] = (_Float16)((vb[k] - mean) * rstd * g[k] + be[k]);
    }
  }
  __syncthreads();

  // ---- Phase 1: QKV = Y @ qkv_wT  (64x384 @ 384x1152), fp16 out to LDS ----
#pragma unroll 1
  for (int u = 0; u < 9; ++u) {
    const int nt = wv * 9 + u;        // 72 N-tiles
    const int ncol = nt * 16;
    const _Float16* bbase = qkvT + (size_t)ncol * DIM;
    v8f ac[4];
#pragma unroll
    for (int mt = 0; mt < 4; ++mt) ac[mt] = vzero8();
#pragma unroll
    for (int kt = 0; kt < 12; ++kt) {
      __builtin_prefetch(bbase + (kt + 1) * 32, 0, 1);
      v16h bf = ldB(bbase + kt * 32, DIM);
#pragma unroll
      for (int mt = 0; mt < 4; ++mt)
        ac[mt] = wmma16(ldA(sYO + (size_t)mt * 16 * DIM + kt * 32, DIM), bf, ac[mt]);
    }
    const float qb = qkv_b[ncol + n15];
    if (ncol < DIM) {                 // Q (pre-scaled)
#pragma unroll
      for (int mt = 0; mt < 4; ++mt)
        stTile(sQ + (size_t)mt * 16 * DIM + ncol, DIM, ac[mt], qb, SCALE);
    } else if (ncol < 2 * DIM) {      // K
#pragma unroll
      for (int mt = 0; mt < 4; ++mt)
        stTile(sK + (size_t)mt * 16 * DIM + (ncol - DIM), DIM, ac[mt], qb, 1.f);
    } else {                          // V, stored transposed for PV B-fragments
      const int nv = ncol - 2 * DIM + n15;
#pragma unroll
      for (int mt = 0; mt < 4; ++mt) {
        v8h p;
#pragma unroll
        for (int v = 0; v < 8; ++v) p[v] = (_Float16)(ac[mt][v] + qb);
        *(v8h*)(sVt + (size_t)nv * NPAD + mt * 16 + mb) = p;
      }
    }
  }
  __syncthreads();

  // ---- Phase 2: scores (QK^T, one 16x16x32 WMMA per tile) + bias + mask +
  //               softmax + PV, per (head, M-tile) unit ----
#pragma unroll 1
  for (int u = 0; u < 6; ++u) {
    const int unit = wv * 6 + u;      // 48 units = 12 heads x 4 M-tiles
    const int h  = unit >> 2;
    const int mt = unit & 3;
    const v16h aq = ldA(sQ + (size_t)mt * 16 * DIM + h * HD, DIM);
    v8f s[4];
#pragma unroll
    for (int t = 0; t < 4; ++t)
      s[t] = wmma16(aq, ldB(sK + (size_t)t * 16 * DIM + h * HD, DIM), vzero8());

    _Float16* Pw = sPW + (size_t)wv * (16 * NPAD);
#pragma unroll
    for (int v = 0; v < 8; ++v) {
      const int r = mt * 16 + mb + v;
      const int ri = r / WS, rj = r % WS;
      const int lr = regid(wy * WS + ri) * 3 + regid(wx * WS + rj);
      float xv[4];
      float mx = -1e30f;
#pragma unroll
      for (int t = 0; t < 4; ++t) {
        const int c = t * 16 + n15;
        float val = -1e30f;
        if (c < NTOK && r < NTOK) {
          const int ci = c / WS, cj = c % WS;
          const int lc = regid(wy * WS + ci) * 3 + regid(wx * WS + cj);
          val = s[t][v] + biasTab[(size_t)(((h << 6) + r) << 6) + c] + ((lr != lc) ? -100.f : 0.f);
        }
        xv[t] = val;
        mx = fmaxf(mx, val);
      }
#pragma unroll
      for (int off = 8; off; off >>= 1) mx = fmaxf(mx, __shfl_xor(mx, off, 32));
      float sm = 0.f, pr[4];
#pragma unroll
      for (int t = 0; t < 4; ++t) {
        pr[t] = (xv[t] > -1e29f) ? __expf(xv[t] - mx) : 0.f;
        sm += pr[t];
      }
#pragma unroll
      for (int off = 8; off; off >>= 1) sm += __shfl_xor(sm, off, 32);
      const float inv = (sm > 0.f) ? (1.f / sm) : 0.f;
#pragma unroll
      for (int t = 0; t < 4; ++t)
        Pw[(size_t)(mb + v) * NPAD + t * 16 + n15] = (_Float16)(pr[t] * inv);
    }

    // PV: O-tile(16 x 32) = P(16x64) @ V(64x32); B from transposed V
#pragma unroll
    for (int dt = 0; dt < 2; ++dt) {
      v8f o = vzero8();
#pragma unroll
      for (int kt = 0; kt < 2; ++kt)
        o = wmma16(ldA(Pw + kt * 32, NPAD),
                   ldB(sVt + (size_t)(h * HD + dt * 16) * NPAD + kt * 32, NPAD), o);
      stTile(sYO + (size_t)mt * 16 * DIM + h * HD + dt * 16, DIM, o, 0.f, 1.f);
    }
  }
  __syncthreads();

  // ---- Phase 3: proj = O @ proj_wT + b, residual add, window-reverse scatter ----
#pragma unroll 1
  for (int u = 0; u < 12; ++u) {
    const int idx = wv * 12 + u;      // 96 tiles = 4 x 24
    const int mt = idx / 24;
    const int nt = idx % 24;
    const int ncol = nt * 16;
    const _Float16* bbase = projT + (size_t)ncol * DIM;
    v8f acc = vzero8();
#pragma unroll
    for (int kt = 0; kt < 12; ++kt)
      acc = wmma16(ldA(sYO + (size_t)mt * 16 * DIM + kt * 32, DIM),
                   ldB(bbase + kt * 32, DIM), acc);
    const int col = ncol + n15;
    const float pb = proj_b[col];
#pragma unroll
    for (int v = 0; v < 8; ++v) {
      const int m = mt * 16 + mb + v;
      if (m < NTOK) {
        int ti = m / WS, tj = m - ti * WS;
        int hh = wy * WS + ti + SS_; if (hh >= H_) hh -= H_;
        int ww = wx * WS + tj + SS_; if (ww >= W_) ww -= W_;
        const size_t off = ((size_t)b * (H_ * W_) + hh * W_ + ww) * DIM + col;
        out[off] = x[off] + acc[v] + pb;
      }
    }
  }
}

// ---------------------------------------------------------------------------
// Kernel 2: fused LN2 + MLP (GEMM1 + exact GELU + GEMM2) + residual, in place
// One workgroup per 64 tokens.
// ---------------------------------------------------------------------------
__global__ __launch_bounds__(256)
void swin_mlp_kernel(const float* __restrict__ ln2_g, const float* __restrict__ ln2_b,
                     const float* __restrict__ b1, const float* __restrict__ b2,
                     const _Float16* __restrict__ w1T, const _Float16* __restrict__ w2T,
                     float* __restrict__ out) {
  __shared__ __align__(16) _Float16 sH[64 * DIM];     // LN2 activations (fp16)
  __shared__ __align__(16) _Float16 sG[64 * HID];     // GELU(h@W1+b1) (fp16)

  const int wv = threadIdx.x >> 5;
  const unsigned lane = threadIdx.x & 31u;
  const int n15 = lane & 15;
  const int mb  = (lane & 16u) ? 8 : 0;
  const size_t T0 = (size_t)blockIdx.x * 64;

  // ---- Async-DMA the x1 tile (64x384 fp32) into LDS (sG region is dead) ----
  float* sX = (float*)sG;                       // 98,304 B staging
  for (int i = threadIdx.x; i < 64 * (DIM / 4); i += 256) {
    const float* g = out + T0 * DIM + (size_t)i * 4;
    async_g2l_b128(sX + (size_t)i * 4, g);
  }
  wait_async0();
  __syncthreads();

  // ---- LN2 from LDS staging ----
  {
    const int c0 = lane * 12;
    float g[12], be[12];
#pragma unroll
    for (int k = 0; k < 12; ++k) { g[k] = ln2_g[c0 + k]; be[k] = ln2_b[c0 + k]; }
    for (int t = wv; t < 64; t += 8) {
      const float* xr = sX + (size_t)t * DIM + c0;
      float vb[12];
#pragma unroll
      for (int q = 0; q < 3; ++q) {
        float4 f = *(const float4*)(xr + q * 4);
        vb[q * 4 + 0] = f.x; vb[q * 4 + 1] = f.y; vb[q * 4 + 2] = f.z; vb[q * 4 + 3] = f.w;
      }
      float s = 0.f, ss = 0.f;
#pragma unroll
      for (int k = 0; k < 12; ++k) { s += vb[k]; ss += vb[k] * vb[k]; }
#pragma unroll
      for (int off = 16; off; off >>= 1) {
        s  += __shfl_xor(s,  off, 32);
        ss += __shfl_xor(ss, off, 32);
      }
      float mean = s * (1.f / DIM);
      float rstd = rsqrtf(ss * (1.f / DIM) - mean * mean + 1e-5f);
#pragma unroll
      for (int k = 0; k < 12; ++k)
        sH[(size_t)t * DIM + c0 + k] = (_Float16)((vb[k] - mean) * rstd * g[k] + be[k]);
    }
  }
  __syncthreads();

  // ---- GEMM1 (64x384 @ 384x1536) + GELU (overwrites sX staging) ----
#pragma unroll 1
  for (int u = 0; u < 48; ++u) {
    const int idx = wv * 48 + u;      // 384 tiles = 4 x 96
    const int mt = idx / 96;
    const int nt = idx % 96;
    const int ncol = nt * 16;
    const _Float16* bbase = w1T + (size_t)ncol * DIM;
    v8f acc = vzero8();
#pragma unroll
    for (int kt = 0; kt < 12; ++kt) {
      __builtin_prefetch(bbase + (kt + 1) * 32, 0, 1);
      acc = wmma16(ldA(sH + (size_t)mt * 16 * DIM + kt * 32, DIM),
                   ldB(bbase + kt * 32, DIM), acc);
    }
    const float bb = b1[ncol + n15];
#pragma unroll
    for (int v = 0; v < 8; ++v) {
      float t2 = acc[v] + bb;
      float ge = 0.5f * t2 * (1.0f + erff(t2 * 0.70710678118654752f));
      sG[(size_t)(mt * 16 + mb + v) * HID + ncol + n15] = (_Float16)ge;
    }
  }
  __syncthreads();

  // ---- GEMM2 (64x1536 @ 1536x384) + residual ----
#pragma unroll 1
  for (int u = 0; u < 12; ++u) {
    const int idx = wv * 12 + u;      // 96 tiles = 4 x 24
    const int mt = idx / 24;
    const int nt = idx % 24;
    const int ncol = nt * 16;
    const _Float16* bbase = w2T + (size_t)ncol * HID;
    v8f acc = vzero8();
#pragma unroll 4
    for (int kt = 0; kt < 48; ++kt) {
      __builtin_prefetch(bbase + (kt + 1) * 32, 0, 1);
      acc = wmma16(ldA(sG + (size_t)mt * 16 * HID + kt * 32, HID),
                   ldB(bbase + kt * 32, HID), acc);
    }
    const float bb = b2[ncol + n15];
#pragma unroll
    for (int v = 0; v < 8; ++v) {
      const size_t off = (T0 + mt * 16 + mb + v) * DIM + ncol + n15;
      out[off] = out[off] + acc[v] + bb;
    }
  }
}

// ---------------------------------------------------------------------------
extern "C" void kernel_launch(void* const* d_in, const int* in_sizes, int n_in,
                              void* d_out, int out_size, void* d_ws, size_t ws_size,
                              hipStream_t stream) {
  (void)in_sizes; (void)n_in; (void)out_size; (void)ws_size;
  const float* x      = (const float*)d_in[0];
  const float* ln1_g  = (const float*)d_in[1];
  const float* ln1_b  = (const float*)d_in[2];
  const float* qkv_w  = (const float*)d_in[3];
  const float* qkv_b  = (const float*)d_in[4];
  const float* relb   = (const float*)d_in[5];
  const float* proj_w = (const float*)d_in[6];
  const float* proj_b = (const float*)d_in[7];
  const float* ln2_g  = (const float*)d_in[8];
  const float* ln2_b  = (const float*)d_in[9];
  const float* mlp_w1 = (const float*)d_in[10];
  const float* mlp_b1 = (const float*)d_in[11];
  const float* mlp_w2 = (const float*)d_in[12];
  const float* mlp_b2 = (const float*)d_in[13];

  char* ws = (char*)d_ws;
  _Float16* qkvT   = (_Float16*)(ws);                 // 1152x384 fp16
  _Float16* projT  = (_Float16*)(ws + 884736);        // 384x384 fp16
  _Float16* w1T    = (_Float16*)(ws + 1179648);       // 1536x384 fp16
  _Float16* w2T    = (_Float16*)(ws + 2359296);       // 384x1536 fp16
  float*    biasTab = (float*)(ws + 3538944);         // 12x64x64 fp32

  const int prep_total = 1152 * 384 + 384 * 384 + 1536 * 384 + 384 * 1536 + NHEAD * NPAD * NPAD;
  swin_prep_kernel<<<(prep_total + 255) / 256, 256, 0, stream>>>(
      qkv_w, proj_w, mlp_w1, mlp_w2, relb, qkvT, projT, w1T, w2T, biasTab);

  swin_attn_kernel<<<64 * 16, 256, 0, stream>>>(
      x, ln1_g, ln1_b, qkv_b, proj_b, qkvT, projT, biasTab, (float*)d_out);

  swin_mlp_kernel<<<(64 * H_ * W_) / 64, 256, 0, stream>>>(
      ln2_g, ln2_b, mlp_b1, mlp_b2, w1T, w2T, (float*)d_out);
}
